// BespokeEmbedding_74397423501937
// MI455X (gfx1250) — matrix operand order repaired and checked by reading, not compiled, not run on hardware
//
#include <hip/hip_runtime.h>
#include <stdint.h>

typedef unsigned int   u32;
typedef unsigned short u16;
typedef __attribute__((ext_vector_type(4)))  u32    v4u;
typedef __attribute__((ext_vector_type(8)))  float  v8f;
typedef __attribute__((ext_vector_type(16))) __bf16 v16bf;

union Frag16 { v4u q[2]; v16bf v; };

__device__ __forceinline__ u16 f2bf(float f) {
  union { float f; u32 u; } x; x.f = f;
  u32 r = x.u + 0x7FFFu + ((x.u >> 16) & 1u);   // round-to-nearest-even
  return (u16)(r >> 16);
}

__global__ void zero_counts_kernel(int* counts) {
  if (threadIdx.x < 4) counts[threadIdx.x] = 0;
}

__global__ void bucket_kernel(const int* __restrict__ token_ids,
                              const int* __restrict__ cat_table,
                              int n_tok, int* __restrict__ counts,
                              int* __restrict__ lists) {
  int t = blockIdx.x * blockDim.x + threadIdx.x;
  if (t >= n_tok) return;
  int tok = token_ids[t];
  int c   = cat_table[tok] & 3;
  int pos = atomicAdd(&counts[c], 1);
  lists[(size_t)c * n_tok + pos] = t;
}

#define BN_CHUNK  128
#define SB_STRIDE 40                    // 32 K-elems + 8 pad (ushorts): 80B rows, 16B aligned
#define SB_ELEMS  (BN_CHUNK * SB_STRIDE)

// One block: 64 gathered tokens x full D output for one category.
// 8 waves: wave&3 -> 16-row M strip, wave>>2 -> 64-col N half of a 128-col chunk.
__global__ void __launch_bounds__(256)
gemm_cat_kernel(const int* __restrict__ list,
                const int* __restrict__ countp,
                const int* __restrict__ token_ids,
                const float* __restrict__ E,
                const float* __restrict__ Wm,
                const float* __restrict__ bias,
                int d, int D,
                float* __restrict__ out) {
  const int nTok     = *countp;
  const int tileBase = blockIdx.x * 64;
  if (tileBase >= nTok) return;

  extern __shared__ char smem_raw[];
  u16* sA = (u16*)smem_raw;                 // 64 rows x (d+8) bf16, row-major
  const int arow = d + 8;                   // (d+8)*2 bytes is 16B-aligned for d%8==0
  u16* sB0 = sA + (size_t)64 * arow;        // [n_local][SB_STRIDE] bf16
  u16* sB1 = sB0 + SB_ELEMS;

  const int tid  = threadIdx.x;
  const int lane = tid & 31;
  const int wave = tid >> 5;
  const int hl   = lane >> 4;               // lane half (K-split per ISA A/B layout)
  const int lm   = lane & 15;
  const int ms   = (wave & 3) * 16;         // M strip base within 64-row tile
  const int nt0  = (wave >> 2) * 4;         // first of 4 N-tiles in the 128-col chunk

  // ---- Stage A: gather 64 embedding rows -> bf16 in LDS (HBM-read once) ----
  for (int m = wave; m < 64; m += 8) {
    int  idx = tileBase + m;
    bool vld = idx < nTok;
    int  t   = vld ? list[idx] : 0;
    const float* src = E + (size_t)token_ids[t] * d;
    u16* dst = sA + (size_t)m * arow;
    for (int k = lane; k < d; k += 32)
      dst[k] = vld ? f2bf(src[k]) : (u16)0;
  }
  __syncthreads();

  const int nsteps  = d >> 5;               // K steps of 32
  const int nchunks = D / BN_CHUNK;

  for (int nc = 0; nc < nchunks; ++nc) {
    const int nbase = nc * BN_CHUNK;
    v8f acc[4] = {};

    // Stage a 32x128 f32 W tile as bf16 into LDS, [n][k] layout.
    auto stageB = [&](u16* sBd, int k0) {
      if (k0 + 32 < d)                      // prefetch next K-chunk rows
        __builtin_prefetch((const void*)&Wm[(size_t)(k0 + 32 + (tid & 31)) * D + nbase], 0, 1);
      for (int i = tid; i < BN_CHUNK * 32; i += 256) {
        int kk = i >> 7;                    // 0..31
        int j  = i & (BN_CHUNK - 1);        // 0..127 (coalesced along N)
        sBd[j * SB_STRIDE + kk] = f2bf(Wm[(size_t)(k0 + kk) * D + nbase + j]);
      }
    };

    stageB(sB0, 0);
    __syncthreads();
    int buf = 0;
    for (int ks = 0; ks < nsteps; ++ks) {
      const int k0 = ks << 5;
      if (ks + 1 < nsteps) stageB(buf ? sB0 : sB1, k0 + 32);   // double-buffer
      const u16* sB = buf ? sB1 : sB0;

      // A fragment: lanes<16 hold K[0..7],K[16..23]; lanes>=16 hold K[8..15],K[24..31]
      Frag16 a;
      const u16* ap = sA + (size_t)(ms + lm) * arow + k0 + hl * 8;
      a.q[0] = *(const v4u*)ap;
      a.q[1] = *(const v4u*)(ap + 16);

#pragma unroll
      for (int t = 0; t < 4; ++t) {
        // B fragment: N = lane%16; lanes<16 hold K[0..7],K[16..23]; lanes>=16 hold K[8..15],K[24..31]
        Frag16 b;
        const u16* bp = sB + ((nt0 + t) * 16 + lm) * SB_STRIDE + hl * 8;
        b.q[0] = *(const v4u*)bp;
        b.q[1] = *(const v4u*)(bp + 16);
        acc[t] = __builtin_amdgcn_wmma_f32_16x16x32_bf16(
            false, a.v, false, b.v, (short)0, acc[t], false, false);
      }
      __syncthreads();
      buf ^= 1;
    }

    // ---- Epilogue: bias + scatter rows via token list ----
#pragma unroll
    for (int t = 0; t < 4; ++t) {
      const int   col = nbase + (nt0 + t) * 16 + lm;
      const float bv  = bias[col];
#pragma unroll
      for (int v = 0; v < 8; ++v) {
        int row = ms + v + hl * 8;          // C layout: VGPR v -> M = v (+8 for upper lanes)
        int idx = tileBase + row;
        if (idx < nTok) {
          int slot = list[idx];             // flat token position
          out[(size_t)slot * D + col] = acc[t][v] + bv;
        }
      }
    }
  }
}

extern "C" void kernel_launch(void* const* d_in, const int* in_sizes, int n_in,
                              void* d_out, int out_size, void* d_ws, size_t ws_size,
                              hipStream_t stream) {
  const int* token_ids = (const int*)d_in[0];
  const int* cat_table = (const int*)d_in[1];
  const int  n_tok = in_sizes[0];           // B*S = 32768
  const int  V     = in_sizes[1];           // 50257

  const float* E[4]; const float* W[4]; const float* Bb[4];
  int dims[4];
  for (int c = 0; c < 4; ++c) {
    E[c]  = (const float*)d_in[2 + 3 * c];
    W[c]  = (const float*)d_in[3 + 3 * c];
    Bb[c] = (const float*)d_in[4 + 3 * c];
    dims[c] = in_sizes[2 + 3 * c] / V;      // 1536,1024,512,256
  }
  const int D = in_sizes[4];                // 1024 (bias length)
  float* out  = (float*)d_out;

  int* counts = (int*)d_ws;                 // [4] counters
  int* lists  = counts + 4;                 // [4][n_tok] token slots

  zero_counts_kernel<<<1, 32, 0, stream>>>(counts);
  bucket_kernel<<<(n_tok + 255) / 256, 256, 0, stream>>>(
      token_ids, cat_table, n_tok, counts, lists);

  // Opt in to >64KB dynamic LDS (CDNA5: 320KB/WGP). Max tile: d=1536 -> ~213KB.
  int maxd = 0;
  for (int c = 0; c < 4; ++c) maxd = dims[c] > maxd ? dims[c] : maxd;
  size_t maxSmem = (size_t)64 * (maxd + 8) * 2 + 2 * (size_t)SB_ELEMS * 2;
  (void)hipFuncSetAttribute(reinterpret_cast<const void*>(gemm_cat_kernel),
                            hipFuncAttributeMaxDynamicSharedMemorySize,
                            (int)maxSmem);

  const int gridX = (n_tok + 63) / 64;      // worst case: all tokens one category
  for (int c = 0; c < 4; ++c) {
    size_t smem = (size_t)64 * (dims[c] + 8) * 2 + 2 * (size_t)SB_ELEMS * 2;
    gemm_cat_kernel<<<dim3(gridX), 256, smem, stream>>>(
        lists + (size_t)c * n_tok, counts + c, token_ids,
        E[c], W[c], Bb[c], dims[c], D, out);
  }
}
